// Olmo3_2HybridGatedDeltaNet_65627100283214
// MI455X (gfx1250) — matrix operand order repaired and checked by reading, not compiled, not run on hardware
//
#include <hip/hip_runtime.h>

// ---------------------------------------------------------------------------
// Olmo3 hybrid GatedDeltaNet forward for gfx1250 (MI455X).
// bf16 WMMA (v_wmma_f32_16x16x32_bf16) GEMMs with async-to-LDS double-buffered
// A-tile staging (global_load_async_to_lds_b128 / s_wait_asynccnt);
// f32 VALU + LDS for the chunked delta-rule scan (192-way dv-slice parallel).
// ---------------------------------------------------------------------------

typedef __attribute__((ext_vector_type(16))) __bf16 bf16x16;
typedef __attribute__((ext_vector_type(8)))  float  f32x8;

#define T_LEN   2048
#define HID     2048
#define NHEAD   16
#define DK      96
#define DV      192
#define KDIM    1536          // NHEAD * DK
#define VDIM    3072          // NHEAD * DV
#define NFUSE   9216          // KDIM + KDIM + VDIM + VDIM (q|k|v|gate)
#define CHUNK   64
#define NCHUNK  32            // T_LEN / CHUNK
#define EPS_F   1e-6f
#define TKB     32            // K elements per WMMA step

__device__ __forceinline__ float sigmoidf_(float x) { return 1.f / (1.f + __expf(-x)); }
__device__ __forceinline__ float siluf_(float x)    { return x * sigmoidf_(x); }

__device__ __forceinline__ unsigned lds_addr(const void* p) {
    // low 32 bits of a generic pointer to __shared__ == wave LDS byte offset
    return (unsigned)(unsigned long long)p;
}

__device__ __forceinline__ void async_copy_b128(unsigned lds, const void* gp) {
    asm volatile("global_load_async_to_lds_b128 %0, %1, off"
                 :: "v"(lds), "v"(gp) : "memory");
}

__device__ __forceinline__ void wait_async0() {
    asm volatile("s_wait_asynccnt 0x0" ::: "memory");
}

// ---------------------------------------------------------------------------
// Conversion kernels
// ---------------------------------------------------------------------------
__global__ void cvt_f32_bf16(const float* __restrict__ x, __bf16* __restrict__ y, int n) {
    int i = blockIdx.x * 256 + threadIdx.x;
    if (i < n) y[i] = (__bf16)x[i];
}

// fused transposed weight: WT[n][k], n in [0,9216): q | k | v | gate
__global__ void build_wt(const float* __restrict__ Wq, const float* __restrict__ Wk,
                         const float* __restrict__ Wv, const float* __restrict__ Wg,
                         __bf16* __restrict__ WT) {
    int i = blockIdx.x * 256 + threadIdx.x;
    if (i >= NFUSE * HID) return;
    int n = i / HID, k = i % HID;
    float v;
    if      (n < KDIM)            v = Wq[(size_t)k * KDIM + n];
    else if (n < 2 * KDIM)        v = Wk[(size_t)k * KDIM + (n - KDIM)];
    else if (n < 2 * KDIM + VDIM) v = Wv[(size_t)k * VDIM + (n - 2 * KDIM)];
    else                          v = Wg[(size_t)k * VDIM + (n - 2 * KDIM - VDIM)];
    WT[(size_t)n * HID + k] = (__bf16)v;
}

// WoT[n][k] = Wo[k][n], Wo is [VDIM][HID]
__global__ void build_wot(const float* __restrict__ Wo, __bf16* __restrict__ WoT) {
    int i = blockIdx.x * 256 + threadIdx.x;
    if (i >= HID * VDIM) return;
    int n = i / VDIM, k = i % VDIM;
    WoT[(size_t)n * VDIM + k] = (__bf16)Wo[(size_t)k * HID + n];
}

// ---------------------------------------------------------------------------
// bf16 WMMA GEMM: C[M][N] = A[M][K] * BT[N][K]^T   (BT row-major N x K)
// Block = 256 threads = 8 waves; block tile 128x128; wave tile 32x64.
// A tile (128x32 bf16 = 8KB) double-buffered in LDS via async-to-LDS DMA.
// Fragment layouts per CDNA5 ISA 7.12.2 (wave32).
// ---------------------------------------------------------------------------
__global__ __launch_bounds__(256)
void wmma_gemm_bf16(const __bf16* __restrict__ A, const __bf16* __restrict__ BT,
                    float* __restrict__ C, int M, int N, int K) {
    __shared__ __align__(16) __bf16 tA[2][128][TKB];   // 2 x 8 KB

    const int tid  = threadIdx.x;
    const int lane = tid & 31;
    const int wave = tid >> 5;
    const int bm   = blockIdx.y * 128;
    const int bn   = blockIdx.x * 128;
    const int wm   = (wave & 3) * 32;     // wave M offset within tile
    const int wn   = (wave >> 2) * 64;    // wave N offset within tile
    const int l16  = lane & 15;
    const int half = lane >> 4;

    // async A-tile loader: 512 16-byte chunks, 2 per thread
    const int c0 = tid * 2;
    auto issue_tile = [&](int k0, int buf) {
#pragma unroll
        for (int q = 0; q < 2; ++q) {
            int c    = c0 + q;
            int row  = c >> 2;          // 4 chunks per 64B row
            int part = c & 3;
            unsigned lds = lds_addr(&tA[buf][row][part * 8]);
            const __bf16* gp = A + (size_t)(bm + row) * K + k0 + part * 8;
            async_copy_b128(lds, gp);
        }
    };

    f32x8 acc[8] = {};   // [mfrag(2)][nfrag(4)]

    const __bf16* pb[4];
#pragma unroll
    for (int f = 0; f < 4; ++f)
        pb[f] = BT + (size_t)(bn + wn + f * 16 + l16) * K;

    issue_tile(0, 0);
    const int ksteps = K / TKB;
    for (int kt = 0; kt < ksteps; ++kt) {
        const int buf = kt & 1;
        wait_async0();          // this wave's tile-kt chunks have landed
        __syncthreads();        // all waves' chunks visible; prior reads done
        if (kt + 1 < ksteps) issue_tile((kt + 1) * TKB, buf ^ 1);

        // A fragments from LDS (two 16x32 frags)
        bf16x16 a0, a1;
        {
            const __bf16* r0 = &tA[buf][wm + l16][0];
            const __bf16* r1 = &tA[buf][wm + 16 + l16][0];
#pragma unroll
            for (int j = 0; j < 8; ++j) {
                a0[j]     = r0[half * 8 + j];
                a0[8 + j] = r0[16 + half * 8 + j];
                a1[j]     = r1[half * 8 + j];
                a1[8 + j] = r1[16 + half * 8 + j];
            }
        }

        const int k0 = kt * TKB;
#pragma unroll
        for (int f = 0; f < 4; ++f) {
            bf16x16 b;
#pragma unroll
            for (int j = 0; j < 16; ++j) b[j] = pb[f][k0 + (j < 8 ? half * 8 + j : 16 + half * 8 + (j - 8))];
            if (kt + 1 < ksteps) __builtin_prefetch(pb[f] + k0 + TKB, 0, 1);
            acc[f]     = __builtin_amdgcn_wmma_f32_16x16x32_bf16(false, a0, false, b,
                                                                 (short)0, acc[f], false, false);
            acc[4 + f] = __builtin_amdgcn_wmma_f32_16x16x32_bf16(false, a1, false, b,
                                                                 (short)0, acc[4 + f], false, false);
        }
        // no trailing barrier needed: ping-pong buffer + top-of-loop barrier
    }

    // C/D layout: VGPR r -> row = base + r + 8*half, col = base + l16
#pragma unroll
    for (int mf = 0; mf < 2; ++mf) {
#pragma unroll
        for (int f = 0; f < 4; ++f) {
            int col = bn + wn + f * 16 + l16;
#pragma unroll
            for (int r = 0; r < 8; ++r) {
                int row = bm + wm + mf * 16 + r + 8 * half;
                C[(size_t)row * N + col] = acc[mf * 4 + f][r];
            }
        }
    }
}

// ---------------------------------------------------------------------------
// Causal depthwise conv (K=4) + SiLU over the q|k|v segments of Ypre
// ---------------------------------------------------------------------------
__global__ void conv_silu(const float* __restrict__ Y,
                          const float* __restrict__ cq, const float* __restrict__ ck,
                          const float* __restrict__ cv,
                          float* __restrict__ q, float* __restrict__ k, float* __restrict__ v) {
    int i = blockIdx.x * 256 + threadIdx.x;
    if (i >= T_LEN * (2 * KDIM + VDIM)) return;
    int t = i / (2 * KDIM + VDIM);
    int c = i % (2 * KDIM + VDIM);
    const float* w;
    float* out;
    int oc, ow;
    if (c < KDIM)            { w = cq + (size_t)c * 4;              out = q; oc = c;            ow = KDIM; }
    else if (c < 2 * KDIM)   { w = ck + (size_t)(c - KDIM) * 4;     out = k; oc = c - KDIM;     ow = KDIM; }
    else                     { w = cv + (size_t)(c - 2 * KDIM) * 4; out = v; oc = c - 2 * KDIM; ow = VDIM; }
    float s = 0.f;
#pragma unroll
    for (int j = 0; j < 4; ++j) {
        int tt = t + j - 3;
        if (tt >= 0) s += Y[(size_t)tt * NFUSE + c] * w[j];
    }
    out[(size_t)t * ow + oc] = siluf_(s);
}

// ---------------------------------------------------------------------------
// beta = 2*sigmoid(h@Wb); g = -exp(A_log)*softplus(h@Wa + dt_bias)
// ---------------------------------------------------------------------------
__global__ __launch_bounds__(256)
void beta_g_kernel(const float* __restrict__ h, const float* __restrict__ Wb,
                   const float* __restrict__ Wa, const float* __restrict__ A_log,
                   const float* __restrict__ dt_bias,
                   float* __restrict__ beta, float* __restrict__ g) {
    int t     = blockIdx.x;
    int grp   = threadIdx.x >> 3;   // 0..31
    int sub   = threadIdx.x & 7;
    int head  = grp & 15;
    int which = grp >> 4;           // 0 = beta, 1 = g
    const float* W = which ? Wa : Wb;
    float s = 0.f;
    for (int kk = sub; kk < HID; kk += 8)
        s += h[(size_t)t * HID + kk] * W[(size_t)kk * NHEAD + head];
    for (int off = 4; off; off >>= 1) s += __shfl_down(s, off, 8);
    if (sub == 0) {
        if (which == 0) {
            beta[t * NHEAD + head] = 2.f * sigmoidf_(s);
        } else {
            float x  = s + dt_bias[head];
            float sp = (x > 20.f) ? x : log1pf(__expf(x));
            g[t * NHEAD + head] = -__expf(A_log[head]) * sp;
        }
    }
}

// ---------------------------------------------------------------------------
// in-place L2 norm of q (scaled by dk^-0.5) and k, per (t, head), one wave each
// ---------------------------------------------------------------------------
__global__ void l2norm_qk(float* __restrict__ q, float* __restrict__ k) {
    int id    = blockIdx.x;            // 0 .. 2*T*NHEAD-1
    int lane  = threadIdx.x;           // 32 threads
    int which = id >= T_LEN * NHEAD;
    int th    = which ? id - T_LEN * NHEAD : id;
    int t = th / NHEAD, h = th % NHEAD;
    float* base = (which ? k : q) + (size_t)t * KDIM + h * DK;
    float x0 = base[lane], x1 = base[lane + 32], x2 = base[lane + 64];
    float ss = x0 * x0 + x1 * x1 + x2 * x2;
    for (int off = 16; off; off >>= 1) ss += __shfl_xor(ss, off, 32);
    float sc = rsqrtf(ss + EPS_F);
    if (!which) sc *= 0.10206207261596577f;   // 96^-0.5
    base[lane]      = x0 * sc;
    base[lane + 32] = x1 * sc;
    base[lane + 64] = x2 * sc;
}

// ---------------------------------------------------------------------------
// Per (head, chunk) prep: A matrix, forward substitution for w and u,
// precomputed attn = (q@k^T)*D, qg = q*e^gc, kdec = k*e^(gl-gc), gl.
// ---------------------------------------------------------------------------
__global__ __launch_bounds__(256)
void dn_prep(const float* __restrict__ qn, const float* __restrict__ kn,
             const float* __restrict__ v,  const float* __restrict__ beta,
             const float* __restrict__ g,
             float* __restrict__ u_g, float* __restrict__ w_g,
             float* __restrict__ attn_g, float* __restrict__ qg_g,
             float* __restrict__ kdec_g, float* __restrict__ gl_g) {
    __shared__ float sA[CHUNK][CHUNK];   // 16 KB
    __shared__ float sB[CHUNK][DK];      // 24 KB (kk, then w, then u-halves)
    __shared__ float gcs[CHUNK], bcs[CHUNK];

    int hn = blockIdx.x;
    int h  = hn >> 5;        // NCHUNK == 32
    int n  = hn & 31;
    int t0 = n * CHUNK;
    int tid = threadIdx.x;

    if (tid == 0) {
        float acc = 0.f;
        for (int i = 0; i < CHUNK; ++i) { acc += g[(t0 + i) * NHEAD + h]; gcs[i] = acc; }
    }
    if (tid < CHUNK) bcs[tid] = beta[(t0 + tid) * NHEAD + h];
    __syncthreads();

    for (int e = tid; e < CHUNK * DK; e += 256) {
        int r = e / DK, d = e % DK;
        sB[r][d] = kn[(size_t)(t0 + r) * KDIM + h * DK + d];
    }
    __syncthreads();

    // A[i][j] = -(b_i k_i . k_j) * exp(gc_i - gc_j), j < i
    for (int e = tid; e < CHUNK * CHUNK; e += 256) {
        int i = e >> 6, j = e & 63;
        float a = 0.f;
        if (j < i) {
            float s = 0.f;
            for (int d = 0; d < DK; ++d) s += sB[i][d] * sB[j][d];
            a = -s * bcs[i] * __expf(gcs[i] - gcs[j]);
        }
        sA[i][j] = a;
    }
    __syncthreads();

    float gl = gcs[CHUNK - 1];
    if (tid == 0) gl_g[hn] = gl;
    size_t b64 = (size_t)hn * CHUNK;

    // attn[i][j] = (q_i . k_j) * exp(gc_i - gc_j), j <= i
    for (int e = tid; e < CHUNK * CHUNK; e += 256) {
        int i = e >> 6, j = e & 63;
        float a = 0.f;
        if (j <= i) {
            const float* qrow = qn + (size_t)(t0 + i) * KDIM + h * DK;
            float s = 0.f;
            for (int d = 0; d < DK; ++d) s += qrow[d] * sB[j][d];
            a = s * __expf(gcs[i] - gcs[j]);
        }
        attn_g[(b64 + i) * CHUNK + j] = a;
    }
    for (int e = tid; e < CHUNK * DK; e += 256) {
        int i = e / DK, d = e % DK;
        const float* qrow = qn + (size_t)(t0 + i) * KDIM + h * DK;
        qg_g[(b64 + i) * DK + d]   = qrow[d] * __expf(gcs[i]);
        kdec_g[(b64 + i) * DK + d] = sB[i][d] * __expf(gl - gcs[i]);
    }
    __syncthreads();

    // w = Tm @ (k*b*e^gc) via forward substitution, in place in sB
    for (int i = 0; i < CHUNK; ++i) {
        for (int d = tid; d < DK; d += 256) {
            float acc = sB[i][d] * bcs[i] * __expf(gcs[i]);
            for (int j = 0; j < i; ++j) acc += sA[i][j] * sB[j][d];
            sB[i][d] = acc;
        }
        __syncthreads();
    }
    for (int e = tid; e < CHUNK * DK; e += 256) {
        int i = e / DK, d = e % DK;
        w_g[(b64 + i) * DK + d] = sB[i][d];
    }
    __syncthreads();

    // u = Tm @ (v*b), two 96-column half passes through sB
    for (int p = 0; p < 2; ++p) {
        for (int e = tid; e < CHUNK * DK; e += 256) {
            int i = e / DK, d = e % DK;
            sB[i][d] = v[(size_t)(t0 + i) * VDIM + h * DV + p * DK + d] * bcs[i];
        }
        __syncthreads();
        for (int i = 1; i < CHUNK; ++i) {
            for (int d = tid; d < DK; d += 256) {
                float acc = sB[i][d];
                for (int j = 0; j < i; ++j) acc += sA[i][j] * sB[j][d];
                sB[i][d] = acc;
            }
            __syncthreads();
        }
        for (int e = tid; e < CHUNK * DK; e += 256) {
            int i = e / DK, d = e % DK;
            u_g[(b64 + i) * DV + p * DK + d] = sB[i][d];
        }
        __syncthreads();
    }
}

// ---------------------------------------------------------------------------
// Sequential scan over chunks; one block per (head, dv-slice of 16):
// 16 heads x 12 slices = 192 independent blocks (dv is fully independent
// in the recurrence given precomputed attn/u/w/qg/kdec).
// ---------------------------------------------------------------------------
#define DVS    16
#define NSLICE 12            // DV / DVS
__global__ __launch_bounds__(256)
void dn_scan(const float* __restrict__ u_g, const float* __restrict__ w_g,
             const float* __restrict__ attn_g, const float* __restrict__ qg_g,
             const float* __restrict__ kdec_g, const float* __restrict__ gl_g,
             float* __restrict__ o_buf) {
    __shared__ float S[DK][DVS];      // 6 KB running state slice
    __shared__ float vn[CHUNK][DVS];  // 4 KB v_new

    int h  = blockIdx.x / NSLICE;
    int sl = blockIdx.x % NSLICE;
    int c0 = sl * DVS;
    int tid = threadIdx.x;

    for (int e = tid; e < DK * DVS; e += 256) S[e / DVS][e % DVS] = 0.f;
    __syncthreads();

    for (int n = 0; n < NCHUNK; ++n) {
        int hn = h * NCHUNK + n;
        size_t b64 = (size_t)hn * CHUNK;
        int t0 = n * CHUNK;

        // v_new = u - w @ S
        for (int e = tid; e < CHUNK * DVS; e += 256) {
            int i = e / DVS, c = e % DVS;
            float acc = u_g[(b64 + i) * DV + c0 + c];
            const float* wrow = w_g + (b64 + i) * DK;
            for (int kk = 0; kk < DK; ++kk) acc -= wrow[kk] * S[kk][c];
            vn[i][c] = acc;
        }
        __syncthreads();

        // o = qg @ S + attn @ v_new
        for (int e = tid; e < CHUNK * DVS; e += 256) {
            int i = e / DVS, c = e % DVS;
            const float* qrow = qg_g + (b64 + i) * DK;
            float acc = 0.f;
            for (int kk = 0; kk < DK; ++kk) acc += qrow[kk] * S[kk][c];
            const float* arow = attn_g + (b64 + i) * CHUNK;
            for (int j = 0; j <= i; ++j) acc += arow[j] * vn[j][c];
            o_buf[(size_t)(t0 + i) * VDIM + h * DV + c0 + c] = acc;
        }
        __syncthreads();

        // S = S * e^gl + kdec^T @ v_new
        float egl = __expf(gl_g[hn]);
        for (int e = tid; e < DK * DVS; e += 256) {
            int kk = e / DVS, c = e % DVS;
            float acc = S[kk][c] * egl;
            for (int i = 0; i < CHUNK; ++i) acc += kdec_g[(b64 + i) * DK + kk] * vn[i][c];
            S[kk][c] = acc;
        }
        __syncthreads();
    }
}

// ---------------------------------------------------------------------------
// o * silu(gate), gated RMSNorm, emit bf16 for the output GEMM; one wave/(t,h)
// ---------------------------------------------------------------------------
__global__ void gate_norm(const float* __restrict__ o_buf, const float* __restrict__ Ypre,
                          const float* __restrict__ nw, __bf16* __restrict__ obf) {
    int th = blockIdx.x;   // T * NHEAD
    int t = th / NHEAD, h = th % NHEAD;
    int lane = threadIdx.x;
    float x[6];
    float ss = 0.f;
#pragma unroll
    for (int j = 0; j < 6; ++j) {
        int c = lane + j * 32;
        float ov  = o_buf[(size_t)t * VDIM + h * DV + c];
        float gv  = Ypre[(size_t)t * NFUSE + 2 * KDIM + VDIM + h * DV + c];
        float val = ov * siluf_(gv);
        x[j] = val;
        ss  += val * val;
    }
    for (int off = 16; off; off >>= 1) ss += __shfl_xor(ss, off, 32);
    float sc = rsqrtf(ss / (float)DV + EPS_F);
#pragma unroll
    for (int j = 0; j < 6; ++j) {
        int c = lane + j * 32;
        obf[(size_t)t * VDIM + h * DV + c] = (__bf16)(x[j] * sc * nw[c]);
    }
}

// ---------------------------------------------------------------------------
// launch
// ---------------------------------------------------------------------------
extern "C" void kernel_launch(void* const* d_in, const int* in_sizes, int n_in,
                              void* d_out, int out_size, void* d_ws, size_t ws_size,
                              hipStream_t stream) {
    (void)in_sizes; (void)n_in; (void)out_size; (void)ws_size;

    const float* hidden  = (const float*)d_in[0];
    const float* Wq      = (const float*)d_in[1];
    const float* Wk      = (const float*)d_in[2];
    const float* Wv      = (const float*)d_in[3];
    const float* Wb      = (const float*)d_in[4];
    const float* Wa      = (const float*)d_in[5];
    const float* Wg      = (const float*)d_in[6];
    const float* Wo      = (const float*)d_in[7];
    const float* conv_q  = (const float*)d_in[8];
    const float* conv_k  = (const float*)d_in[9];
    const float* conv_v  = (const float*)d_in[10];
    const float* A_log   = (const float*)d_in[11];
    const float* dt_bias = (const float*)d_in[12];
    const float* norm_w  = (const float*)d_in[13];
    float* out = (float*)d_out;

    char* ws = (char*)d_ws;
    size_t off = 0;
    auto alloc = [&](size_t bytes) -> char* {
        char* p = ws + off;
        off += (bytes + 255) & ~(size_t)255;
        return p;
    };

    __bf16* hbf  = (__bf16*)alloc((size_t)T_LEN * HID * 2);
    __bf16* WT   = (__bf16*)alloc((size_t)NFUSE * HID * 2);
    __bf16* WoT  = (__bf16*)alloc((size_t)HID * VDIM * 2);
    float*  Ypre = (float*) alloc((size_t)T_LEN * NFUSE * 4);
    float*  qb   = (float*) alloc((size_t)T_LEN * KDIM * 4);
    float*  kb   = (float*) alloc((size_t)T_LEN * KDIM * 4);
    float*  vb   = (float*) alloc((size_t)T_LEN * VDIM * 4);
    float*  beta = (float*) alloc((size_t)T_LEN * NHEAD * 4);
    float*  gbuf = (float*) alloc((size_t)T_LEN * NHEAD * 4);
    float*  u_g  = (float*) alloc((size_t)NHEAD * NCHUNK * CHUNK * DV * 4);
    float*  w_g  = (float*) alloc((size_t)NHEAD * NCHUNK * CHUNK * DK * 4);
    float*  at_g = (float*) alloc((size_t)NHEAD * NCHUNK * CHUNK * CHUNK * 4);
    float*  qg_g = (float*) alloc((size_t)NHEAD * NCHUNK * CHUNK * DK * 4);
    float*  kd_g = (float*) alloc((size_t)NHEAD * NCHUNK * CHUNK * DK * 4);
    float*  gl_g = (float*) alloc((size_t)NHEAD * NCHUNK * 4);
    float*  o_b  = (float*) alloc((size_t)T_LEN * VDIM * 4);
    __bf16* obf  = (__bf16*)alloc((size_t)T_LEN * VDIM * 2);

    // 1. conversions
    cvt_f32_bf16<<<(T_LEN * HID) / 256, 256, 0, stream>>>(hidden, hbf, T_LEN * HID);
    build_wt<<<(NFUSE * HID) / 256, 256, 0, stream>>>(Wq, Wk, Wv, Wg, WT);
    build_wot<<<(HID * VDIM) / 256, 256, 0, stream>>>(Wo, WoT);

    // 2. fused input projections: Ypre[T][9216] = hidden @ [Wq|Wk|Wv|Wg]
    wmma_gemm_bf16<<<dim3(NFUSE / 128, T_LEN / 128), 256, 0, stream>>>(
        hbf, WT, Ypre, T_LEN, NFUSE, HID);

    // 3. conv + silu -> q, k, v
    conv_silu<<<(T_LEN * (2 * KDIM + VDIM)) / 256, 256, 0, stream>>>(
        Ypre, conv_q, conv_k, conv_v, qb, kb, vb);

    // 4. beta / g heads
    beta_g_kernel<<<T_LEN, 256, 0, stream>>>(hidden, Wb, Wa, A_log, dt_bias, beta, gbuf);

    // 5. l2 norm q (scaled) and k
    l2norm_qk<<<2 * T_LEN * NHEAD, 32, 0, stream>>>(qb, kb);

    // 6. per-(head, chunk) prep
    dn_prep<<<NHEAD * NCHUNK, 256, 0, stream>>>(qb, kb, vb, beta, gbuf,
                                                u_g, w_g, at_g, qg_g, kd_g, gl_g);

    // 7. sequential scan per (head, dv-slice)
    dn_scan<<<NHEAD * NSLICE, 256, 0, stream>>>(u_g, w_g, at_g, qg_g, kd_g, gl_g, o_b);

    // 8. gate + RMSNorm -> bf16
    gate_norm<<<T_LEN * NHEAD, 32, 0, stream>>>(o_b, Ypre, norm_w, obf);

    // 9. output projection: out = obf @ Wo  (via WoT)
    wmma_gemm_bf16<<<dim3(HID / 128, T_LEN / 128), 256, 0, stream>>>(
        obf, WoT, out, T_LEN, HID, VDIM);
}
